// TestAllReduceRMSNormStaticQuantFP8Model_9947144257742
// MI455X (gfx1250) — 8-bit matrix operand order repaired and checked by reading, compiled, not run on hardware
//
#include <hip/hip_runtime.h>


#define KT 8192
#define KH 4096

typedef __attribute__((ext_vector_type(16))) int   v16i;
typedef __attribute__((ext_vector_type(8)))  float v8f;

typedef int v4i_g __attribute__((vector_size(16)));
typedef __attribute__((address_space(1))) v4i_g* as1_v4i_p;
typedef __attribute__((address_space(3))) v4i_g* as3_v4i_p;

#if __has_builtin(__builtin_amdgcn_global_load_async_to_lds_b128)
#define HAS_ASYNC_LDS 1
#endif

__device__ __forceinline__ unsigned char f32_to_e4m3(float x) {
    unsigned int u    = __float_as_uint(x);
    unsigned int sign = (u >> 24) & 0x80u;
    unsigned int au   = u & 0x7FFFFFFFu;
    if (au > 0x43E00000u) au = 0x43E00000u;
    if (au < 0x3C000000u) {
        float f = __uint_as_float(au);
        int   m = (int)rintf(f * 512.0f);
        return (unsigned char)(sign | (unsigned int)m);
    }
    unsigned int lsb = (au >> 20) & 1u;
    unsigned int r   = au + 0x0007FFFFu + lsb;
    unsigned int e   = (r >> 23) - 127u + 7u;
    unsigned int m   = (r >> 20) & 7u;
    return (unsigned char)(sign | (e << 3) | m);
}

__global__ void quant_w_kernel(const float* __restrict__ w,
                               unsigned char* __restrict__ qwT) {
    size_t i = (size_t)blockIdx.x * blockDim.x + threadIdx.x;
    size_t l   = i / ((size_t)KH * KH);
    size_t rem = i % ((size_t)KH * KH);
    size_t k   = rem / KH;
    size_t n   = rem % KH;
    qwT[l * (size_t)KH * KH + n * (size_t)KH + k] = f32_to_e4m3(w[i]);
}

__global__ void relu_kernel(const float* __restrict__ h, float* __restrict__ o) {
    size_t i = ((size_t)blockIdx.x * blockDim.x + threadIdx.x) * 4;
    float4 v = *(const float4*)(h + i);
    v.x = fmaxf(v.x, 0.0f); v.y = fmaxf(v.y, 0.0f);
    v.z = fmaxf(v.z, 0.0f); v.w = fmaxf(v.w, 0.0f);
    *(float4*)(o + i) = v;
}

__global__ void __launch_bounds__(256)
rms_quant_kernel(const float* __restrict__ x, const float* __restrict__ g,
                 const float* __restrict__ s_ptr, unsigned char* __restrict__ q) {
    __shared__ float red[256];
    const int row = blockIdx.x, tid = threadIdx.x;
    const float* xr = x + (size_t)row * KH;
    float4 v[4];
    float  ss = 0.0f;
#pragma unroll
    for (int i = 0; i < 4; ++i) {
        v[i] = *(const float4*)(xr + (tid + 256 * i) * 4);
        ss += v[i].x * v[i].x + v[i].y * v[i].y + v[i].z * v[i].z + v[i].w * v[i].w;
    }
    red[tid] = ss;
    __syncthreads();
    for (int st = 128; st > 0; st >>= 1) {
        if (tid < st) red[tid] += red[tid + st];
        __syncthreads();
    }
    const float rinv  = rsqrtf(red[0] * (1.0f / KH) + 1e-6f);
    const float inv_s = 1.0f / s_ptr[0];
#pragma unroll
    for (int i = 0; i < 4; ++i) {
        const int    e  = (tid + 256 * i) * 4;
        const float4 gv = *(const float4*)(g + e);
        uchar4 qb;
        qb.x = f32_to_e4m3(v[i].x * rinv * gv.x * inv_s);
        qb.y = f32_to_e4m3(v[i].y * rinv * gv.y * inv_s);
        qb.z = f32_to_e4m3(v[i].z * rinv * gv.z * inv_s);
        qb.w = f32_to_e4m3(v[i].w * rinv * gv.w * inv_s);
        *(uchar4*)(q + (size_t)row * KH + e) = qb;
    }
}

__global__ void __launch_bounds__(256)
rms_out_kernel(const float* __restrict__ x, const float* __restrict__ g,
               float* __restrict__ out) {
    __shared__ float red[256];
    const int row = blockIdx.x, tid = threadIdx.x;
    const float* xr = x + (size_t)row * KH;
    float4 v[4];
    float  ss = 0.0f;
#pragma unroll
    for (int i = 0; i < 4; ++i) {
        v[i] = *(const float4*)(xr + (tid + 256 * i) * 4);
        ss += v[i].x * v[i].x + v[i].y * v[i].y + v[i].z * v[i].z + v[i].w * v[i].w;
    }
    red[tid] = ss;
    __syncthreads();
    for (int st = 128; st > 0; st >>= 1) {
        if (tid < st) red[tid] += red[tid + st];
        __syncthreads();
    }
    const float rinv = rsqrtf(red[0] * (1.0f / KH) + 1e-6f);
#pragma unroll
    for (int i = 0; i < 4; ++i) {
        const int    e  = (tid + 256 * i) * 4;
        const float4 gv = *(const float4*)(g + e);
        float4 o;
        o.x = v[i].x * rinv * gv.x; o.y = v[i].y * rinv * gv.y;
        o.z = v[i].z * rinv * gv.z; o.w = v[i].w * rinv * gv.w;
        *(float4*)(out + (size_t)row * KH + e) = o;
    }
}

#define LDA 144

__device__ __forceinline__ v16i load_a_frag(const unsigned char* As, int mrow, int lane) {
    const int m = lane & 15, hi = lane >> 4;
    const unsigned char* base = As + (size_t)(mrow + m) * LDA;
    union { int i[16]; v16i v; } f;
#pragma unroll
    for (int half = 0; half < 2; ++half)
#pragma unroll
        for (int p = 0; p < 4; ++p) {
            const int2 d = *(const int2*)(base + 64 * half + 16 * p + 8 * hi);
            f.i[half * 8 + 2 * p]     = d.x;
            f.i[half * 8 + 2 * p + 1] = d.y;
        }
    return f.v;
}

__device__ __forceinline__ v16i load_b_frag(const unsigned char* Bs, int nrow, int lane) {
    const int n = lane & 15, hi = lane >> 4;
    const unsigned char* base = Bs + (size_t)(nrow + n) * LDA;
    union { int i[16]; v16i v; } f;
#pragma unroll
    for (int half = 0; half < 2; ++half)
#pragma unroll
        for (int p = 0; p < 4; ++p) {
            const int2 d = *(const int2*)(base + 64 * half + 16 * p + 8 * hi);
            f.i[half * 8 + 2 * p]     = d.x;
            f.i[half * 8 + 2 * p + 1] = d.y;
        }
    return f.v;
}

__device__ __forceinline__ void stage_tile(const unsigned char* __restrict__ g,
                                           unsigned char* l, int tid) {
#if defined(HAS_ASYNC_LDS)
#pragma unroll
    for (int u = 0; u < 4; ++u) {
        const int idx4 = tid + 256 * u;
        const int r = idx4 >> 3, c = idx4 & 7;
        __builtin_amdgcn_global_load_async_to_lds_b128(
            (as1_v4i_p)(g + (size_t)r * KH + c * 16),
            (as3_v4i_p)(l + (size_t)r * LDA + c * 16),
            0, 0);
    }
#else
    uint4 t[4];
#pragma unroll
    for (int u = 0; u < 4; ++u) {
        const int idx4 = tid + 256 * u;
        const int r = idx4 >> 3, c = idx4 & 7;
        t[u] = *(const uint4*)(g + (size_t)r * KH + c * 16);
    }
#pragma unroll
    for (int u = 0; u < 4; ++u) {
        const int idx4 = tid + 256 * u;
        const int r = idx4 >> 3, c = idx4 & 7;
        *(uint4*)(l + (size_t)r * LDA + c * 16) = t[u];
    }
#endif
}

__device__ __forceinline__ void wait_async_stage() {
#if defined(HAS_ASYNC_LDS)
#if __has_builtin(__builtin_amdgcn_s_wait_asynccnt)
    __builtin_amdgcn_s_wait_asynccnt(0);
#else
    asm volatile("s_wait_asynccnt 0x0" ::: "memory");
#endif
#endif
}

__global__ void __launch_bounds__(256)
gemm_fp8_kernel(const unsigned char* __restrict__ qx,
                const unsigned char* __restrict__ qwT,
                const float* __restrict__ s_in,
                const float* __restrict__ s_w,
                const float* __restrict__ resid,
                float* __restrict__ out) {
    __shared__ __align__(16) unsigned char As[2][128 * LDA];
    __shared__ __align__(16) unsigned char Bs[2][128 * LDA];

    const int tid  = threadIdx.x;
    const int lane = tid & 31;
    const int wave = tid >> 5;
    const int wm   = wave >> 1;
    const int wn   = wave & 1;
    const int bm   = blockIdx.y * 128;
    const int bn   = blockIdx.x * 128;
    const float s  = s_in[0] * s_w[0];

    const unsigned char* gA = qx  + (size_t)bm * KH;
    const unsigned char* gB = qwT + (size_t)bn * KH;

    v8f acc[2][4];
#pragma unroll
    for (int i = 0; i < 2; ++i)
#pragma unroll
        for (int j = 0; j < 4; ++j)
            acc[i][j] = (v8f){0.f, 0.f, 0.f, 0.f, 0.f, 0.f, 0.f, 0.f};

    stage_tile(gA, &As[0][0], tid);
    stage_tile(gB, &Bs[0][0], tid);
    wait_async_stage();
    __syncthreads();

    for (int kk = 0; kk < KH; kk += 128) {
        const int cur = (kk >> 7) & 1;

        if (kk + 128 < KH) {
            stage_tile(gA + kk + 128, &As[cur ^ 1][0], tid);
            stage_tile(gB + kk + 128, &Bs[cur ^ 1][0], tid);
        }

        const v16i a0 = load_a_frag(&As[cur][0], wm * 32,      lane);
        const v16i a1 = load_a_frag(&As[cur][0], wm * 32 + 16, lane);
#pragma unroll
        for (int j = 0; j < 4; ++j) {
            const v16i b = load_b_frag(&Bs[cur][0], wn * 64 + j * 16, lane);
            acc[0][j] = __builtin_amdgcn_wmma_f32_16x16x128_fp8_fp8(
                a0, b, (short)0, acc[0][j], false, false);
            acc[1][j] = __builtin_amdgcn_wmma_f32_16x16x128_fp8_fp8(
                a1, b, (short)0, acc[1][j], false, false);
        }

        wait_async_stage();
        __syncthreads();
    }

    const int hi = lane >> 4, nl = lane & 15;
#pragma unroll
    for (int i = 0; i < 2; ++i)
#pragma unroll
        for (int j = 0; j < 4; ++j) {
            const int gcol = bn + wn * 64 + j * 16 + nl;
#pragma unroll
            for (int r = 0; r < 8; ++r) {
                const int grow = bm + wm * 32 + i * 16 + hi * 8 + r;
                const size_t o = (size_t)grow * KH + gcol;
                out[o] = acc[i][j][r] * s + resid[o];
            }
        }
}

extern "C" void kernel_launch(void* const* d_in, const int* in_sizes, int n_in,
                              void* d_out, int out_size, void* d_ws, size_t ws_size,
                              hipStream_t stream) {
    (void)in_sizes; (void)n_in; (void)out_size; (void)ws_size;
    const float* h   = (const float*)d_in[0];
    const float* nw  = (const float*)d_in[1];
    const float* w   = (const float*)d_in[2];
    const float* wsc = (const float*)d_in[3];
    const float* sc  = (const float*)d_in[4];
    float* out = (float*)d_out;

    char* wsb = (char*)d_ws;
    const size_t HH = (size_t)KH * KH;
    const size_t TH = (size_t)KT * KH;
    unsigned char* qwT  = (unsigned char*)wsb;
    unsigned char* qx   = (unsigned char*)(wsb + 3 * HH);
    float*         bufA = (float*)(wsb + 3 * HH + TH);
    float*         bufB = bufA + TH;

    const dim3 gemmGrid(KH / 128, KT / 128);

    quant_w_kernel<<<(unsigned)(3 * HH / 256), 256, 0, stream>>>(w, qwT);

    relu_kernel<<<(unsigned)(TH / 4 / 256), 256, 0, stream>>>(h, bufA);

    rms_quant_kernel<<<KT, 256, 0, stream>>>(bufA, nw + 0 * KH, sc + 0, qx);
    gemm_fp8_kernel<<<gemmGrid, 256, 0, stream>>>(qx, qwT + 0 * HH, sc + 0, wsc + 0, bufA, bufB);

    rms_quant_kernel<<<KT, 256, 0, stream>>>(bufB, nw + 1 * KH, sc + 1, qx);
    gemm_fp8_kernel<<<gemmGrid, 256, 0, stream>>>(qx, qwT + 1 * HH, sc + 1, wsc + 1, bufB, bufA);

    rms_quant_kernel<<<KT, 256, 0, stream>>>(bufA, nw + 2 * KH, sc + 2, qx);
    gemm_fp8_kernel<<<gemmGrid, 256, 0, stream>>>(qx, qwT + 2 * HH, sc + 2, wsc + 2, bufA, bufB);

    rms_out_kernel<<<KT, 256, 0, stream>>>(bufB, nw + 3 * KH, out);
}
